// InfiniAttention_75445395521790
// MI455X (gfx1250) — matrix operand order repaired and checked by reading, not compiled, not run on hardware
//
#include <hip/hip_runtime.h>
#include <hip/hip_bf16.h>
#include <math.h>

#define BB  2
#define SS  2048
#define HH  2048
#define NHH 16
#define HDD 128

typedef __bf16 bf16t;
typedef __attribute__((ext_vector_type(16))) __bf16 v16bf;
typedef __attribute__((ext_vector_type(8)))  float  v8f;

union BFragU { v16bf v; float4 q[2]; };

__device__ inline v8f v8f_zero() {
  v8f z = {0.f,0.f,0.f,0.f,0.f,0.f,0.f,0.f};
  return z;
}

__device__ inline v8f wmma_bf16(v16bf a, v16bf b, v8f c) {
  // (neg_a, A, neg_b, B, c_mod, C, reuse_a, reuse_b)
  return __builtin_amdgcn_wmma_f32_16x16x32_bf16(false, a, false, b, (short)0, c, false, false);
}

// A fragment (16x32 bf16, M=lane&15): element i -> k = ((i>=8)?16:0) + 8*(lane>>4) + (i&7)
__device__ inline v16bf frag_a_rm(const bf16t* A, int ld, int row, int kbase, int lane) {
  const bf16t* p = A + (size_t)row * ld + kbase + ((lane >> 4) << 3);
  BFragU f;
  f.q[0] = *(const float4*)(p);
  f.q[1] = *(const float4*)(p + 16);
  return f.v;
}

// B fragment (32x16 bf16, N=lane&15): element i -> k = ((i>=8)?16:0) + 8*(lane>>4) + (i&7) ; needs B^T row-major
__device__ inline v16bf frag_b_cm(const bf16t* Bt, int ld, int col, int kbase, int lane) {
  const bf16t* p = Bt + (size_t)col * ld + kbase + ((lane >> 4) << 3);
  BFragU f;
  f.q[0] = *(const float4*)(p);
  f.q[1] = *(const float4*)(p + 16);
  return f.v;
}

// ---------------- elementwise / layout kernels ----------------

__global__ __launch_bounds__(256) void cvt_f32_bf16_kernel(const float* __restrict__ in,
                                                           bf16t* __restrict__ out, size_t n) {
  size_t i = (size_t)blockIdx.x * 256 + threadIdx.x;
  if (i < n) out[i] = (bf16t)in[i];
}

// LDS-tiled transpose: out[z][c][r] = (bf16)in[b*bstride + h*hstride + r*rs + c]
// 64x64 tile per block; coalesced reads and writes. R, C multiples of 64 or 128.
template <typename TIN>
__global__ __launch_bounds__(256) void transpose_tiled_kernel(const TIN* __restrict__ in,
                                                              bf16t* __restrict__ out,
                                                              int R, int C, long rs,
                                                              long bstride, long hstride, int nh) {
  __shared__ bf16t tile[64][74];  // pad: 37-dword row stride -> conflict-free
  int z = blockIdx.z;
  int b = z / nh, h = z - b * nh;
  size_t ibase = (size_t)b * bstride + (size_t)h * hstride;
  size_t obase = (size_t)z * (size_t)R * C;
  int r0 = blockIdx.x * 64, c0 = blockIdx.y * 64;
  int x = threadIdx.x & 63, y0 = threadIdx.x >> 6;

  const TIN* ip = in + ibase + (size_t)r0 * rs + c0;
#pragma unroll
  for (int i = 0; i < 16; i++) {
    int y = y0 + i * 4;
    tile[x][y] = (bf16t)ip[(size_t)y * rs + x];  // transposed store into LDS
  }
  __syncthreads();
  bf16t* op = out + obase + (size_t)c0 * R + r0;
#pragma unroll
  for (int i = 0; i < 16; i++) {
    int y = y0 + i * 4;
    op[(size_t)y * R + x] = tile[y][x];
  }
}

__device__ inline bf16t elu1_bf16(float x) {
  return (bf16t)((x > 0.f) ? (x + 1.f) : __expf(x));
}

// RoPE + sigma; all outputs row-major [bh,s,d] (fully coalesced).
__global__ __launch_bounds__(256) void rope_post_kernel(const bf16t* __restrict__ Qg,
                                                        const bf16t* __restrict__ Kg,
                                                        bf16t* __restrict__ Qb,
                                                        bf16t* __restrict__ Kb,
                                                        bf16t* __restrict__ SQb,
                                                        bf16t* __restrict__ SKg) {
  size_t t = (size_t)blockIdx.x * 256 + threadIdx.x;  // t < B*S*NH*64 = 2^22
  int j = (int)(t & 63);
  int h = (int)((t >> 6) & (NHH - 1));
  int s = (int)((t >> 10) & (SS - 1));
  int b = (int)(t >> 21);

  size_t base = ((size_t)b * SS + s) * HH + (size_t)h * HDD;

  float q1 = (float)Qg[base + j];
  float q2 = (float)Qg[base + j + 64];
  float k1 = (float)Kg[base + j];
  float k2 = (float)Kg[base + j + 64];

  float inv = __powf(10000.f, -(float)j * (1.f / 64.f));
  float ang = (float)s * inv;
  float sn, cs;
  __sincosf(ang, &sn, &cs);

  float q1r = q1 * cs - q2 * sn;
  float q2r = q2 * cs + q1 * sn;
  float k1r = k1 * cs - k2 * sn;
  float k2r = k2 * cs + k1 * sn;

  size_t bh = (size_t)b * NHH + h;
  size_t o1 = (bh * SS + s) * HDD + j;
  size_t o2 = o1 + 64;
  Qb[o1] = (bf16t)q1r;  Qb[o2] = (bf16t)q2r;
  Kb[o1] = (bf16t)k1r;  Kb[o2] = (bf16t)k2r;
  SQb[o1] = elu1_bf16(q1r);
  SQb[o2] = elu1_bf16(q2r);
  SKg[o1] = elu1_bf16(k1r);
  SKg[o2] = elu1_bf16(k2r);
}

__global__ __launch_bounds__(256) void combine_kernel(const float* __restrict__ ctx,
                                                      const float* __restrict__ mo,
                                                      const float* __restrict__ beta,
                                                      bf16t* __restrict__ Ob) {
  size_t t = (size_t)blockIdx.x * 256 + threadIdx.x;  // t < B*S*H
  int d = (int)(t & (HDD - 1));
  int h = (int)((t >> 7) & (NHH - 1));
  int s = (int)((t >> 11) & (SS - 1));
  int b = (int)(t >> 22);
  size_t src = (((size_t)b * NHH + h) * SS + s) * HDD + d;
  float bs = 1.f / (1.f + __expf(-beta[0]));
  Ob[t] = (bf16t)(bs * ctx[src] + (1.f - bs) * mo[src]);
}

// ---------------- WMMA GEMM: C[M,N] = A[M,K] * B[K,N], B given as B^T[N,K] ----------------
// Block = 256 threads = 8 waves in 2(m) x 4(n); block tile 128x256; wave tile 64x64.
template <bool F32OUT>
__global__ __launch_bounds__(256) void gemm_bf16_kernel(const bf16t* __restrict__ A,
                                                        const bf16t* __restrict__ Bt,
                                                        void* __restrict__ C,
                                                        int M, int N, int K) {
  int lane = threadIdx.x & 31;
  int wave = threadIdx.x >> 5;
  int wm = wave & 1, wn = wave >> 1;
  int bm = blockIdx.y * 128 + wm * 64;
  int bn = blockIdx.x * 256 + wn * 64;

  v8f acc[4][4];
#pragma unroll
  for (int i = 0; i < 4; i++)
#pragma unroll
    for (int j = 0; j < 4; j++) acc[i][j] = v8f_zero();

  for (int ks = 0; ks < K; ks += 32) {
    v16bf af[4], bfv[4];
#pragma unroll
    for (int i = 0; i < 4; i++)
      af[i] = frag_a_rm(A, K, bm + i * 16 + (lane & 15), ks, lane);
#pragma unroll
    for (int j = 0; j < 4; j++)
      bfv[j] = frag_b_cm(Bt, K, bn + j * 16 + (lane & 15), ks, lane);
#pragma unroll
    for (int i = 0; i < 4; i++)
#pragma unroll
      for (int j = 0; j < 4; j++)
        acc[i][j] = wmma_bf16(af[i], bfv[j], acc[i][j]);
  }

#pragma unroll
  for (int i = 0; i < 4; i++)
#pragma unroll
    for (int j = 0; j < 4; j++)
#pragma unroll
      for (int r = 0; r < 8; r++) {
        int row = bm + i * 16 + r + ((lane >> 4) << 3);
        int col = bn + j * 16 + (lane & 15);
        if (F32OUT) ((float*)C)[(size_t)row * N + col] = acc[i][j][r];
        else        ((bf16t*)C)[(size_t)row * N + col] = (bf16t)acc[i][j][r];
      }
}

// ---------------- per-wave flash attention ----------------
__global__ __launch_bounds__(256) void attn_kernel(const bf16t* __restrict__ Qb,
                                                   const bf16t* __restrict__ Kb,
                                                   const bf16t* __restrict__ Vt,
                                                   float* __restrict__ ctx) {
  __shared__ __align__(16) bf16t lds_p[8][16][32];

  int lane = threadIdx.x & 31;
  int wave = threadIdx.x >> 5;
  int bh = blockIdx.x >> 4;   // S/128 = 16 q-blocks per head
  int qb = blockIdx.x & 15;
  int q0 = qb * 128 + wave * 16;

  const bf16t* Qh = Qb + (size_t)bh * SS * HDD;
  const bf16t* Kh = Kb + (size_t)bh * SS * HDD;
  const bf16t* Vh = Vt + (size_t)bh * HDD * SS;

  v16bf qf[4];
#pragma unroll
  for (int c = 0; c < 4; c++)
    qf[c] = frag_a_rm(Qh, HDD, q0 + (lane & 15), c * 32, lane);

  float mi[8], li[8];
#pragma unroll
  for (int r = 0; r < 8; r++) { mi[r] = -3.0e38f; li[r] = 0.f; }
  v8f o[8];
#pragma unroll
  for (int e = 0; e < 8; e++) o[e] = v8f_zero();

  const float scale = 0.08838834764831845f;  // 1/sqrt(128)
  int kend = q0 + 16;

  for (int kb = 0; kb < kend; kb += 32) {
    v8f s0 = v8f_zero(), s1 = v8f_zero();
#pragma unroll
    for (int c = 0; c < 4; c++) {
      v16bf b0 = frag_b_cm(Kh, HDD, kb + (lane & 15), c * 32, lane);
      v16bf b1 = frag_b_cm(Kh, HDD, kb + 16 + (lane & 15), c * 32, lane);
      s0 = wmma_bf16(qf[c], b0, s0);
      s1 = wmma_bf16(qf[c], b1, s1);
    }
#pragma unroll
    for (int r = 0; r < 8; r++) { s0[r] *= scale; s1[r] *= scale; }

    if (kb + 31 >= q0) {  // diagonal region: causal mask
      int key0 = kb + (lane & 15);
#pragma unroll
      for (int r = 0; r < 8; r++) {
        int mg = q0 + r + ((lane >> 4) << 3);
        if (key0 > mg)      s0[r] = -1.0e30f;
        if (key0 + 16 > mg) s1[r] = -1.0e30f;
      }
    }

    float mx[8];
#pragma unroll
    for (int r = 0; r < 8; r++) mx[r] = fmaxf(s0[r], s1[r]);
#pragma unroll
    for (int off = 1; off < 16; off <<= 1)
#pragma unroll
      for (int r = 0; r < 8; r++) mx[r] = fmaxf(mx[r], __shfl_xor(mx[r], off, 32));

    float al[8];
#pragma unroll
    for (int r = 0; r < 8; r++) {
      float nm = fmaxf(mi[r], mx[r]);
      al[r] = __expf(mi[r] - nm);
      mi[r] = nm;
    }
#pragma unroll
    for (int r = 0; r < 8; r++) {
      s0[r] = __expf(s0[r] - mi[r]);
      s1[r] = __expf(s1[r] - mi[r]);
    }
    float sm[8];
#pragma unroll
    for (int r = 0; r < 8; r++) sm[r] = s0[r] + s1[r];
#pragma unroll
    for (int off = 1; off < 16; off <<= 1)
#pragma unroll
      for (int r = 0; r < 8; r++) sm[r] += __shfl_xor(sm[r], off, 32);
#pragma unroll
    for (int r = 0; r < 8; r++) li[r] = li[r] * al[r] + sm[r];

    // transpose P (C layout -> A layout) through per-wave LDS tile
#pragma unroll
    for (int r = 0; r < 8; r++) {
      int m = r + ((lane >> 4) << 3);
      lds_p[wave][m][lane & 15]        = (bf16t)s0[r];
      lds_p[wave][m][16 + (lane & 15)] = (bf16t)s1[r];
    }
    __builtin_amdgcn_wave_barrier();
    asm volatile("s_wait_dscnt 0" ::: "memory");
    v16bf pf = frag_a_rm(&lds_p[wave][0][0], 32, lane & 15, 0, lane);
    __builtin_amdgcn_wave_barrier();

#pragma unroll
    for (int e = 0; e < 8; e++)
#pragma unroll
      for (int r = 0; r < 8; r++) o[e][r] *= al[r];
#pragma unroll
    for (int e = 0; e < 8; e++) {
      v16bf vf = frag_b_cm(Vh, SS, e * 16 + (lane & 15), kb, lane);
      o[e] = wmma_bf16(pf, vf, o[e]);
    }
  }

#pragma unroll
  for (int e = 0; e < 8; e++)
#pragma unroll
    for (int r = 0; r < 8; r++) {
      int row = q0 + r + ((lane >> 4) << 3);
      ctx[((size_t)bh * SS + row) * HDD + e * 16 + (lane & 15)] = o[e][r] / li[r];
    }
}

// ---------------- memory read: mo[bh,s,e] = sigma_q[s,:] . M[:,e] ----------------
__global__ __launch_bounds__(256) void memread_kernel(const bf16t* __restrict__ SQb,
                                                      const bf16t* __restrict__ Mtb,
                                                      float* __restrict__ mo) {
  int lane = threadIdx.x & 31;
  int wave = threadIdx.x >> 5;
  int bh = blockIdx.x >> 4;
  int sb = blockIdx.x & 15;
  int s0 = sb * 128 + wave * 16;

  const bf16t* SQh = SQb + (size_t)bh * SS * HDD;
  const bf16t* Mth = Mtb + (size_t)bh * HDD * HDD;

  v8f acc[8];
#pragma unroll
  for (int e = 0; e < 8; e++) acc[e] = v8f_zero();

#pragma unroll
  for (int c = 0; c < 4; c++) {
    v16bf af = frag_a_rm(SQh, HDD, s0 + (lane & 15), c * 32, lane);
#pragma unroll
    for (int e = 0; e < 8; e++) {
      v16bf bfv = frag_b_cm(Mth, HDD, e * 16 + (lane & 15), c * 32, lane);
      acc[e] = wmma_bf16(af, bfv, acc[e]);
    }
  }
#pragma unroll
  for (int e = 0; e < 8; e++)
#pragma unroll
    for (int r = 0; r < 8; r++) {
      int row = s0 + r + ((lane >> 4) << 3);
      mo[((size_t)bh * SS + row) * HDD + e * 16 + (lane & 15)] = acc[e][r];
    }
}

// ---------------- memory update: out2[bh,d,e] = M[bh,d,e] + sum_s sk[s,d] v[s,e] ----------------
__global__ __launch_bounds__(256) void memupd_kernel(const bf16t* __restrict__ SKt,
                                                     const bf16t* __restrict__ Vt,
                                                     const float* __restrict__ Mem,
                                                     float* __restrict__ out2) {
  int lane = threadIdx.x & 31;
  int wave = threadIdx.x >> 5;
  int bh = blockIdx.x;
  int wm = wave & 1, wn = wave >> 1;

  const bf16t* SKh = SKt + (size_t)bh * HDD * SS;
  const bf16t* Vh  = Vt  + (size_t)bh * HDD * SS;

  v8f acc[4][2];
#pragma unroll
  for (int i = 0; i < 4; i++)
#pragma unroll
    for (int j = 0; j < 2; j++) acc[i][j] = v8f_zero();

  for (int ks = 0; ks < SS; ks += 32) {
    v16bf af[4], bfv[2];
#pragma unroll
    for (int i = 0; i < 4; i++)
      af[i] = frag_a_rm(SKh, SS, wm * 64 + i * 16 + (lane & 15), ks, lane);
#pragma unroll
    for (int j = 0; j < 2; j++)
      bfv[j] = frag_b_cm(Vh, SS, wn * 32 + j * 16 + (lane & 15), ks, lane);
#pragma unroll
    for (int i = 0; i < 4; i++)
#pragma unroll
      for (int j = 0; j < 2; j++)
        acc[i][j] = wmma_bf16(af[i], bfv[j], acc[i][j]);
  }

#pragma unroll
  for (int i = 0; i < 4; i++)
#pragma unroll
    for (int j = 0; j < 2; j++)
#pragma unroll
      for (int r = 0; r < 8; r++) {
        int d = wm * 64 + i * 16 + r + ((lane >> 4) << 3);
        int e = wn * 32 + j * 16 + (lane & 15);
        size_t idx = ((size_t)bh * HDD + d) * HDD + e;
        out2[idx] = acc[i][j][r] + Mem[idx];
      }
}

// ---------------- host launch ----------------

extern "C" void kernel_launch(void* const* d_in, const int* in_sizes, int n_in,
                              void* d_out, int out_size, void* d_ws, size_t ws_size,
                              hipStream_t stream) {
  const float* hidden = (const float*)d_in[0];
  const float* memin  = (const float*)d_in[1];
  const float* Wq     = (const float*)d_in[2];
  const float* Wk     = (const float*)d_in[3];
  const float* Wv     = (const float*)d_in[4];
  const float* Wo     = (const float*)d_in[5];
  const float* beta   = (const float*)d_in[6];
  float* out  = (float*)d_out;
  float* out2 = out + (size_t)BB * SS * HH;

  char* w = (char*)d_ws;
  auto alloc = [&](size_t bytes) {
    char* p = w;
    w += (bytes + 255) & ~(size_t)255;
    return p;
  };
  const size_t XB = (size_t)BB * SS * HH;
  const size_t WB = (size_t)HH * HH;
  const size_t MB = (size_t)BB * NHH * HDD * HDD;

  bf16t* Xb  = (bf16t*)alloc(XB * 2);
  bf16t* Wqt = (bf16t*)alloc(WB * 2);
  bf16t* Wkt = (bf16t*)alloc(WB * 2);
  bf16t* Wvt = (bf16t*)alloc(WB * 2);
  bf16t* Wot = (bf16t*)alloc(WB * 2);
  bf16t* Mtb = (bf16t*)alloc(MB * 2);
  bf16t* Qg  = (bf16t*)alloc(XB * 2);
  bf16t* Kg  = (bf16t*)alloc(XB * 2);
  bf16t* Vg  = (bf16t*)alloc(XB * 2);
  bf16t* Qb  = (bf16t*)alloc(XB * 2);
  bf16t* Kb  = (bf16t*)alloc(XB * 2);
  bf16t* SQb = (bf16t*)alloc(XB * 2);
  bf16t* SKg = (bf16t*)alloc(XB * 2);
  bf16t* SKt = (bf16t*)alloc(XB * 2);
  bf16t* Vt  = (bf16t*)alloc(XB * 2);
  float* ctx = (float*)alloc(XB * 4);
  float* mo  = (float*)alloc(XB * 4);
  bf16t* Ob  = (bf16t*)alloc(XB * 2);
  if ((size_t)(w - (char*)d_ws) > ws_size) return;  // workspace too small

  // 1. conversions / transposes (tiled, coalesced)
  cvt_f32_bf16_kernel<<<(int)(XB / 256), 256, 0, stream>>>(hidden, Xb, XB);
  transpose_tiled_kernel<float><<<dim3(32, 32, 1), 256, 0, stream>>>(Wq, Wqt, HH, HH, HH, 0, 0, 1);
  transpose_tiled_kernel<float><<<dim3(32, 32, 1), 256, 0, stream>>>(Wk, Wkt, HH, HH, HH, 0, 0, 1);
  transpose_tiled_kernel<float><<<dim3(32, 32, 1), 256, 0, stream>>>(Wv, Wvt, HH, HH, HH, 0, 0, 1);
  transpose_tiled_kernel<float><<<dim3(32, 32, 1), 256, 0, stream>>>(Wo, Wot, HH, HH, HH, 0, 0, 1);
  transpose_tiled_kernel<float><<<dim3(2, 2, BB * NHH), 256, 0, stream>>>(
      memin, Mtb, HDD, HDD, HDD, (long)HDD * HDD, 0, 1);

  // 2. QKV projections (wave tile 64x64, block tile 128x256)
  dim3 ggrid(HH / 256, (BB * SS) / 128);
  gemm_bf16_kernel<false><<<ggrid, 256, 0, stream>>>(Xb, Wqt, Qg, BB * SS, HH, HH);
  gemm_bf16_kernel<false><<<ggrid, 256, 0, stream>>>(Xb, Wkt, Kg, BB * SS, HH, HH);
  gemm_bf16_kernel<false><<<ggrid, 256, 0, stream>>>(Xb, Wvt, Vg, BB * SS, HH, HH);

  // 3. RoPE / sigma (row-major outputs), then tiled head-transposes for SK and V
  size_t rthreads = (size_t)BB * SS * NHH * 64;
  rope_post_kernel<<<(int)(rthreads / 256), 256, 0, stream>>>(Qg, Kg, Qb, Kb, SQb, SKg);
  transpose_tiled_kernel<bf16t><<<dim3(SS / 64, 2, BB * NHH), 256, 0, stream>>>(
      SKg, SKt, SS, HDD, HDD, (long)SS * HDD, 0, 1);
  transpose_tiled_kernel<bf16t><<<dim3(SS / 64, 2, BB * NHH), 256, 0, stream>>>(
      Vg, Vt, SS, HDD, HH, (long)SS * HH, HDD, NHH);

  // 4. flash attention ; 5. memory read ; 6. memory update
  attn_kernel<<<BB * NHH * (SS / 128), 256, 0, stream>>>(Qb, Kb, Vt, ctx);
  memread_kernel<<<BB * NHH * (SS / 128), 256, 0, stream>>>(SQb, Mtb, mo);
  memupd_kernel<<<BB * NHH, 256, 0, stream>>>(SKt, Vt, memin, out2);

  // 7. combine ; 8. output projection
  combine_kernel<<<(int)(XB / 256), 256, 0, stream>>>(ctx, mo, beta, Ob);
  gemm_bf16_kernel<true><<<ggrid, 256, 0, stream>>>(Ob, Wot, out, BB * SS, HH, HH);
}
